// SimilarityMatrix_30313879175403
// MI455X (gfx1250) — matrix operand order repaired and checked by reading, not compiled, not run on hardware
//
#include <hip/hip_runtime.h>

typedef __attribute__((ext_vector_type(16))) _Float16 v16h;
typedef __attribute__((ext_vector_type(8)))  float    v8f;

#define TM 128           // output rows per block
#define TN 64            // output cols per block
#define TK 32            // K chunk (one f16 WMMA K-step)
#define LDK (TK + 8)     // padded LDS row stride: 40 halves = 80B = 20 banks
                         // -> 16B-aligned fragment segments (ds_load_b128) and
                         //    conflict-free: {20k mod 64} distinct for k=0..15,
                         //    16 lanes x 4 banks tile all 64 banks exactly.

// ---------------------------------------------------------------------------
// wave-per-row dot: out[row] = dot(X[row, 0:D], w[0:D])   (wave32)
// ---------------------------------------------------------------------------
__global__ void rowdot_kernel(const float* __restrict__ X,
                              const float* __restrict__ w,
                              float* __restrict__ out,
                              int nrows, int D) {
    const int wid  = (blockIdx.x * blockDim.x + threadIdx.x) >> 5;
    const int lane = threadIdx.x & 31;
    if (wid >= nrows) return;
    const float* row = X + (size_t)wid * D;
    float s = 0.f;
    for (int d = lane; d < D; d += 32)          // coalesced across the wave
        s += row[d] * w[d];
    #pragma unroll
    for (int off = 16; off > 0; off >>= 1)
        s += __shfl_down(s, off, 32);
    if (lane == 0) out[wid] = s;
}

// ---------------------------------------------------------------------------
// Batched GEMM with split-f16 precision recovery + rank-1 epilogue.
// out[b,t,j] = sum_d (ctx[b,t,d]*wm[d]) * qst[b,j,d] + cterm[b,t] + qterm[b,j]
// ---------------------------------------------------------------------------
__global__ __launch_bounds__(256)
void simmat_kernel(const float* __restrict__ ctx,   // [B,T,D]
                   const float* __restrict__ qst,   // [B,J,D]
                   const float* __restrict__ wker,  // [3*D] (wc|wq|wm)
                   const float* __restrict__ cterm, // [B*T]
                   const float* __restrict__ qterm, // [B*J]
                   float* __restrict__ out,         // [B,T,J]
                   int T, int J, int D) {
    __shared__ alignas(16) _Float16 sAhi[TM][LDK];
    __shared__ alignas(16) _Float16 sAlo[TM][LDK];
    __shared__ alignas(16) _Float16 sBhi[TN][LDK];
    __shared__ alignas(16) _Float16 sBlo[TN][LDK];

    const int b      = blockIdx.z;
    const int rowBlk = blockIdx.y * TM;
    const int colBlk = blockIdx.x * TN;

    const int tid   = threadIdx.x;
    const int wave  = tid >> 5;
    const int lane  = tid & 31;
    const int wm_   = wave & 3;       // wave row-group  (4 x 32 rows)
    const int wn_   = wave >> 2;      // wave col-group  (2 x 32 cols)
    const int mHalf = lane >> 4;      // lane-half per ISA fragment layout
    const int mIdx  = lane & 15;

    const float* Ab = ctx + ((size_t)b * T + rowBlk) * D;
    const float* Bb = qst + ((size_t)b * J + colBlk) * D;
    const float* wm = wker + 2 * D;

    v8f acc[2][2];
    #pragma unroll
    for (int i = 0; i < 2; ++i)
        #pragma unroll
        for (int j = 0; j < 2; ++j)
            acc[i][j] = (v8f){0.f,0.f,0.f,0.f,0.f,0.f,0.f,0.f};

    const int nChunks = D / TK;
    for (int kc = 0; kc < nChunks; ++kc) {
        const int k0 = kc * TK;

        // Prefetch next K chunk (this thread's first segment) into caches.
        if (kc + 1 < nChunks) {
            const int linA = tid;                       // float4 index
            __builtin_prefetch(Ab + (size_t)(linA >> 3) * D + k0 + TK + (linA & 7) * 4, 0, 3);
            const int linB = tid;
            if (linB < (TN * TK) / 4)
                __builtin_prefetch(Bb + (size_t)(linB >> 3) * D + k0 + TK + (linB & 7) * 4, 0, 3);
        }

        // ---- A tile: TM x TK fp32, scale by wm, split into hi/lo f16 ------
        #pragma unroll
        for (int i = 0; i < (TM * TK) / (256 * 4); ++i) {   // 4 float4 / thread
            const int lin = tid + i * 256;                  // float4 index
            const int r   = lin >> 3;                       // TK/4 = 8 float4/row
            const int c4  = lin & 7;
            const float4 av = *(const float4*)(Ab + (size_t)r * D + k0 + c4 * 4);
            const float4 wv = *(const float4*)(wm + k0 + c4 * 4);
            const float vx[4] = {av.x * wv.x, av.y * wv.y, av.z * wv.z, av.w * wv.w};
            #pragma unroll
            for (int e = 0; e < 4; ++e) {
                const _Float16 hi = (_Float16)vx[e];
                const _Float16 lo = (_Float16)(vx[e] - (float)hi);
                sAhi[r][c4 * 4 + e] = hi;
                sAlo[r][c4 * 4 + e] = lo;
            }
        }
        // ---- B tile: TN x TK fp32, split into hi/lo f16 --------------------
        #pragma unroll
        for (int i = 0; i < (TN * TK) / (256 * 4); ++i) {   // 2 float4 / thread
            const int lin = tid + i * 256;
            const int r   = lin >> 3;
            const int c4  = lin & 7;
            const float4 bv = *(const float4*)(Bb + (size_t)r * D + k0 + c4 * 4);
            const float vx[4] = {bv.x, bv.y, bv.z, bv.w};
            #pragma unroll
            for (int e = 0; e < 4; ++e) {
                const _Float16 hi = (_Float16)vx[e];
                const _Float16 lo = (_Float16)(vx[e] - (float)hi);
                sBhi[r][c4 * 4 + e] = hi;
                sBlo[r][c4 * 4 + e] = lo;
            }
        }
        __syncthreads();

        // ---- build fragments per ISA 16-bit A/B VGPR layouts ---------------
        // A 16x32: lanes 0-15 row M=lane, K {0..7,16..23}; lanes 16-31 K {8..15,24..31}
        v16h ahi[2], alo[2], bhi[2], blo[2];
        #pragma unroll
        for (int i = 0; i < 2; ++i) {
            const int r = wm_ * 32 + i * 16 + mIdx;
            #pragma unroll
            for (int e = 0; e < 8; ++e) {
                ahi[i][e]     = sAhi[r][e + mHalf * 8];
                ahi[i][e + 8] = sAhi[r][e + 16 + mHalf * 8];
                alo[i][e]     = sAlo[r][e + mHalf * 8];
                alo[i][e + 8] = sAlo[r][e + 16 + mHalf * 8];
            }
        }
        // B 32x16: lanes 0-15 col N=lane, K {0..7,16..23}; lanes 16-31 K {8..15,24..31}
        #pragma unroll
        for (int j = 0; j < 2; ++j) {
            const int c = wn_ * 32 + j * 16 + mIdx;
            #pragma unroll
            for (int e = 0; e < 8; ++e) {
                bhi[j][e]     = sBhi[c][e + mHalf * 8];
                bhi[j][e + 8] = sBhi[c][e + 16 + mHalf * 8];
                blo[j][e]     = sBlo[c][e + mHalf * 8];
                blo[j][e + 8] = sBlo[c][e + 16 + mHalf * 8];
            }
        }

        // ---- 3-term split-f16 MMA: Alo*Bhi + Ahi*Blo + Ahi*Bhi -------------
        #pragma unroll
        for (int i = 0; i < 2; ++i)
            #pragma unroll
            for (int j = 0; j < 2; ++j) {
                acc[i][j] = __builtin_amdgcn_wmma_f32_16x16x32_f16(
                    false, alo[i], false, bhi[j], (short)0, acc[i][j], false, false);
                acc[i][j] = __builtin_amdgcn_wmma_f32_16x16x32_f16(
                    false, ahi[i], false, blo[j], (short)0, acc[i][j], false, false);
                acc[i][j] = __builtin_amdgcn_wmma_f32_16x16x32_f16(
                    false, ahi[i], false, bhi[j], (short)0, acc[i][j], false, false);
            }
        __syncthreads();
    }

    // ---- epilogue: + c_term[row] + q_term[col], per C-matrix lane layout ---
    // C 16x16 f32: VGPR r -> (M = r + 8*half, N = lane&15)
    const float* ct = cterm + (size_t)b * T + rowBlk;
    const float* qt = qterm + (size_t)b * J + colBlk;
    float* ob = out + ((size_t)b * T + rowBlk) * J + colBlk;
    #pragma unroll
    for (int i = 0; i < 2; ++i)
        #pragma unroll
        for (int j = 0; j < 2; ++j) {
            const int col = wn_ * 32 + j * 16 + mIdx;
            const float qv = qt[col];
            #pragma unroll
            for (int r = 0; r < 8; ++r) {
                const int row = wm_ * 32 + i * 16 + mHalf * 8 + r;
                ob[(size_t)row * J + col] = acc[i][j][r] + ct[row] + qv;
            }
        }
}

// ---------------------------------------------------------------------------
extern "C" void kernel_launch(void* const* d_in, const int* in_sizes, int n_in,
                              void* d_out, int out_size, void* d_ws, size_t ws_size,
                              hipStream_t stream) {
    const float* ctx  = (const float*)d_in[0];
    const float* qst  = (const float*)d_in[1];
    const float* wker = (const float*)d_in[2];

    const long s0 = in_sizes[0], s1 = in_sizes[1], s2 = in_sizes[2];
    const int  D  = (int)(s2 / 3);
    const long BT = s0 / D;
    const long BJ = s1 / D;
    const long B  = (BT * BJ) / (long)out_size;   // (B*T)(B*J)/(B*T*J) = B
    const int  T  = (int)(BT / B);
    const int  J  = (int)(BJ / B);

    float* cterm = (float*)d_ws;       // B*T floats
    float* qterm = cterm + BT;         // B*J floats

    // c_term = context @ wc ; q_term = question @ wq   (wave-per-row, wave32)
    {
        const int rowsPerBlk = 256 / 32;
        rowdot_kernel<<<dim3((unsigned)((BT + rowsPerBlk - 1) / rowsPerBlk)), dim3(256), 0, stream>>>(
            ctx, wker, cterm, (int)BT, D);
        rowdot_kernel<<<dim3((unsigned)((BJ + rowsPerBlk - 1) / rowsPerBlk)), dim3(256), 0, stream>>>(
            qst, wker + D, qterm, (int)BJ, D);
    }

    dim3 grid(J / TN, T / TM, (unsigned)B);       // (8, 16, 32)
    simmat_kernel<<<grid, dim3(256), 0, stream>>>(ctx, qst, wker, cterm, qterm,
                                                  (float*)d_out, T, J, D);
}
